// Seq2SeqAttention_72318659330588
// MI455X (gfx1250) — compile-verified, matrix operand order checked by reading, not run on hardware
//
#include <hip/hip_runtime.h>
#include <hip/hip_bf16.h>
#include <stdint.h>

#define Bb 16
#define Ss 64
#define Ee 512
#define Hh 1024
#define VOo 32000
#define TSTEPS 63   // T-1 decode steps

typedef __attribute__((ext_vector_type(16))) __bf16 v16bf;
typedef __attribute__((ext_vector_type(8)))  __bf16 v8bf;
typedef __attribute__((ext_vector_type(8)))  float  v8f;

// ---------- helpers ----------
__device__ __forceinline__ __bf16 f2bf(float f) {
    unsigned u = __builtin_bit_cast(unsigned, f);
    unsigned r = (u + 0x7FFFu + ((u >> 16) & 1u)) >> 16;
    unsigned short s = (unsigned short)r;
    return __builtin_bit_cast(__bf16, s);
}
__device__ __forceinline__ float sigm(float x) { return 1.0f / (1.0f + __expf(-x)); }

// Fragment loader for V_WMMA_F32_16X16X32_BF16, row-pointer form.
// Lane pair (l, l+16) holds row l; e0..7 = K[kb+8*half .. +7], e8..15 = K[kb+16+8*half .. +7].
__device__ __forceinline__ v16bf frag_row(const __bf16* rowp, int kb, int half) {
    const __bf16* p = rowp + kb + (half << 3);
    v8bf a = *(const v8bf*)p;
    v8bf b = *(const v8bf*)(p + 16);
    v16bf r;
#pragma unroll
    for (int i = 0; i < 8; ++i) { r[i] = a[i]; r[i + 8] = b[i]; }
    return r;
}
__device__ __forceinline__ v8f wmma_bf16(v16bf a, v16bf b, v8f c) {
    return __builtin_amdgcn_wmma_f32_16x16x32_bf16(false, a, false, b, (short)0, c, false, false);
}

// ---------- one-time prep kernels ----------
__global__ void cvt_kernel(const float* __restrict__ s, __bf16* __restrict__ d, long n4) {
    long i = (long)blockIdx.x * blockDim.x + threadIdx.x;
    long stride = (long)gridDim.x * blockDim.x;
    for (; i < n4; i += stride) {
        float4 v = ((const float4*)s)[i];
        __bf16 o[4] = { f2bf(v.x), f2bf(v.y), f2bf(v.z), f2bf(v.w) };
        *(uint64_t*)&d[i * 4] = *(const uint64_t*)o;
    }
}
__global__ void bias_sum_kernel(const float* __restrict__ a, const float* __restrict__ b,
                                float* __restrict__ o, int n) {
    int i = blockIdx.x * blockDim.x + threadIdx.x;
    if (i < n) o[i] = a[i] + b[i];
}
__global__ void embed_kernel(const int* __restrict__ tok, int tokStride,
                             const float* __restrict__ emb, __bf16* __restrict__ dst, int steps) {
    int total = steps * Bb * (Ee / 4);
    int i = blockIdx.x * blockDim.x + threadIdx.x;
    if (i >= total) return;
    int e4 = i % (Ee / 4);
    int r = i / (Ee / 4);
    int b = r % Bb, s = r / Bb;
    int t = tok[b * tokStride + s];
    float4 v = ((const float4*)(emb + (long)t * Ee))[e4];
    __bf16 o[4] = { f2bf(v.x), f2bf(v.y), f2bf(v.z), f2bf(v.w) };
    *(uint64_t*)&dst[((long)(s * Bb + b) * Ee) + e4 * 4] = *(const uint64_t*)o;
}

// ---------- fused LSTM step: gate GEMM (WMMA) + elementwise ----------
// grid = 64 (hidden tiles), block = 128 (4 waves = gates i,f,g,o for this hidden slice)
__global__ void lstm_step_kernel(const __bf16* __restrict__ xt,   // [16][512] bf16
                                 const __bf16* __restrict__ hin,  // [16][1024] bf16
                                 const __bf16* __restrict__ Wih,  // [4096][512]
                                 const __bf16* __restrict__ Whh,  // [4096][1024]
                                 const float*  __restrict__ bias, // [4096] (b_ih+b_hh)
                                 float*        __restrict__ cst,  // [16][1024]
                                 __bf16*       __restrict__ hout, // [16][1024] ping-pong
                                 float* outF, long strideF,       // optional f32 h sink
                                 __bf16* outB, long strideB)      // optional bf16 h sink
{
    __shared__ __align__(16) __bf16 sA[16 * 1536]; // x||h, 48KB
    __shared__ float sG[4 * 256];                  // 4 gate tiles 16x16
    const int tid = threadIdx.x;
    const int wave = tid >> 5, lane = tid & 31, l = lane & 15, half = lane >> 4;
    const int hbase = blockIdx.x * 16;

    // stage A = [x_t | h] into LDS (16B chunks)
    for (int c = tid; c < 1024; c += 128) {             // x: 16*512/8
        int m = c >> 6, o = (c & 63) * 8;
        *(uint4*)&sA[m * 1536 + o] = *(const uint4*)&xt[m * Ee + o];
    }
    for (int c = tid; c < 2048; c += 128) {             // h: 16*1024/8
        int m = c >> 7, o = (c & 127) * 8;
        *(uint4*)&sA[m * 1536 + Ee + o] = *(const uint4*)&hin[m * Hh + o];
    }
    __syncthreads();

    const int n0g = wave * Hh + hbase;                  // global gate row
    const __bf16* arow = sA + l * 1536;
    const __bf16* wihr = Wih + (long)(n0g + l) * Ee;
    const __bf16* whhr = Whh + (long)(n0g + l) * Hh;

    v8f acc0 = {}, acc1 = {};
    // x part: K = 512 (8 x 64), two independent accumulator chains
#pragma unroll
    for (int kb = 0; kb < Ee; kb += 64) {
        v16bf a0 = frag_row(arow, kb, half),      b0 = frag_row(wihr, kb, half);
        v16bf a1 = frag_row(arow, kb + 32, half), b1 = frag_row(wihr, kb + 32, half);
        acc0 = wmma_bf16(a0, b0, acc0);
        acc1 = wmma_bf16(a1, b1, acc1);
    }
    // h part: K = 1024 (16 x 64)
#pragma unroll
    for (int kb = 0; kb < Hh; kb += 64) {
        v16bf a0 = frag_row(arow + Ee, kb, half),      b0 = frag_row(whhr, kb, half);
        v16bf a1 = frag_row(arow + Ee, kb + 32, half), b1 = frag_row(whhr, kb + 32, half);
        acc0 = wmma_bf16(a0, b0, acc0);
        acc1 = wmma_bf16(a1, b1, acc1);
    }
    v8f acc = acc0 + acc1;
    float bv = bias[n0g + l];
#pragma unroll
    for (int r = 0; r < 8; ++r)
        sG[wave * 256 + (r + 8 * half) * 16 + l] = acc[r] + bv;
    __syncthreads();

    for (int idx = tid; idx < 256; idx += 128) {
        int m = idx >> 4, n = idx & 15;
        int hid = hbase + n;
        float gi = sG[idx], gf = sG[256 + idx], gg = sG[512 + idx], go = sG[768 + idx];
        float co = cst[m * Hh + hid];
        float cn = sigm(gf) * co + sigm(gi) * tanhf(gg);
        float hv = sigm(go) * tanhf(cn);
        cst[m * Hh + hid] = cn;
        hout[m * Hh + hid] = f2bf(hv);
        if (outF) outF[(long)m * strideF + hid] = hv;
        if (outB) outB[(long)m * strideB + hid] = f2bf(hv);
    }
}

// ---------- generic small WMMA GEMM: C = A @ W^T + bias ----------
// block = 128 (4 waves, 1 N-tile each); grid.x = Ntiles/4, grid.y = Mtiles
__global__ void gemm16_kernel(const __bf16* __restrict__ A, int lda,
                              const __bf16* __restrict__ W, int K,
                              const float* __restrict__ bias,
                              float* __restrict__ C, int ldc)
{
    const int tid = threadIdx.x;
    const int wave = tid >> 5, lane = tid & 31, l = lane & 15, half = lane >> 4;
    const int mt = blockIdx.y;
    const int nt = blockIdx.x * (blockDim.x >> 5) + wave;
    const int n0 = nt * 16;
    const __bf16* arow = A + ((long)mt * 16 + l) * lda;
    const __bf16* wrow = W + (long)(n0 + l) * K;
    v8f acc0 = {}, acc1 = {};
#pragma unroll 4
    for (int kb = 0; kb < K; kb += 64) {
        v16bf a0 = frag_row(arow, kb, half),      b0 = frag_row(wrow, kb, half);
        v16bf a1 = frag_row(arow, kb + 32, half), b1 = frag_row(wrow, kb + 32, half);
        acc0 = wmma_bf16(a0, b0, acc0);
        acc1 = wmma_bf16(a1, b1, acc1);
    }
    v8f acc = acc0 + acc1;
    float bb = bias[n0 + l];
#pragma unroll
    for (int r = 0; r < 8; ++r)
        C[(long)(mt * 16 + r + 8 * half) * ldc + n0 + l] = acc[r] + bb;
}

// ---------- attention: scores -> softmax -> context (per batch row) ----------
__global__ void attention_kernel(const float* __restrict__ q,      // [16][1024]
                                 const float* __restrict__ Ue,     // [16*64][1024]
                                 const float* __restrict__ va,     // [1024]
                                 const float* __restrict__ bv,     // [1]
                                 const float* __restrict__ enc,    // [16][64][1024]
                                 __bf16* __restrict__ cat)         // step base [16][2048]; write +1024
{
    __shared__ float sQ[Hh];
    __shared__ float sS[Ss];
    const int b = blockIdx.x, tid = threadIdx.x;
    for (int i = tid; i < Hh; i += 256) sQ[i] = q[b * Hh + i];
    __syncthreads();
    const int wave = tid >> 5, lane = tid & 31;
    for (int j = 0; j < 8; ++j) {
        int s = wave * 8 + j;
        const float* ue = Ue + ((long)b * Ss + s) * Hh;
        float sum = 0.f;
        for (int k = lane; k < Hh; k += 32) sum += tanhf(ue[k] + sQ[k]) * va[k];
#pragma unroll
        for (int o = 16; o > 0; o >>= 1) sum += __shfl_xor(sum, o, 32);
        if (lane == 0) sS[s] = sum + bv[0];
    }
    __syncthreads();
    if (tid == 0) {
        float mx = sS[0];
        for (int s = 1; s < Ss; ++s) mx = fmaxf(mx, sS[s]);
        float tot = 0.f;
        for (int s = 0; s < Ss; ++s) { float e = __expf(sS[s] - mx); sS[s] = e; tot += e; }
        float inv = 1.f / tot;
        for (int s = 0; s < Ss; ++s) sS[s] *= inv;
    }
    __syncthreads();
#pragma unroll
    for (int ci = 0; ci < 4; ++ci) {
        int col = tid + ci * 256;
        float acc = 0.f;
        for (int s = 0; s < Ss; ++s)
            acc += sS[s] * enc[((long)b * Ss + s) * Hh + col];
        cat[b * 2048 + 1024 + col] = f2bf(acc);
    }
}

// ---------- batched output projection: [63*16, 2048] @ fc_W^T -> out[b][t][v] ----------
// block = 256 (8 waves); each wave owns 2 N-tiles (32 cols) with 4 independent WMMA chains
// and double-buffered fragments. WG covers 256 cols. grid.x = 125 N-panels, grid.y = 7 M-chunks.
__global__ void fc_gemm_kernel(const __bf16* __restrict__ Acat, // [63][16][2048]
                               const __bf16* __restrict__ Wfc,  // [32000][2048]
                               const float* __restrict__ bias,  // [32000]
                               float* __restrict__ out)         // [16][63][32000]
{
    __shared__ __align__(16) __bf16 sA[16 * 2048]; // 64KB A tile
    const int tid = threadIdx.x;
    const int wave = tid >> 5, lane = tid & 31, l = lane & 15, half = lane >> 4;
    const int n0 = blockIdx.x * 256 + wave * 32;   // first tile cols n0..n0+15, second n0+16..n0+31
    const int m0 = blockIdx.y * 9;
    const int m1 = (m0 + 9 < TSTEPS) ? m0 + 9 : TSTEPS;
    const __bf16* w0 = Wfc + (long)(n0 + l) * 2048;
    const __bf16* w1 = Wfc + (long)(n0 + 16 + l) * 2048;
    const __bf16* arow = sA + l * 2048;

    for (int mt = m0; mt < m1; ++mt) {
        for (int ch = tid; ch < 4096; ch += 256) {   // 16*2048/8
            int m = ch >> 8, o = (ch & 255) * 8;
            *(uint4*)&sA[m * 2048 + o] = *(const uint4*)&Acat[((long)mt * 16 + m) * 2048 + o];
        }
        __syncthreads();

        v8f a00 = {}, a01 = {}, a10 = {}, a11 = {};
        // prologue: fragments for kb = 0
        v16bf fa0 = frag_row(arow, 0, half),  fa1 = frag_row(arow, 32, half);
        v16bf fb00 = frag_row(w0, 0, half),   fb01 = frag_row(w0, 32, half);
        v16bf fb10 = frag_row(w1, 0, half),   fb11 = frag_row(w1, 32, half);
#pragma unroll 2
        for (int kb = 0; kb < 2048; kb += 64) {
            int kn = kb + 64;
            int kc = (kn < 2048) ? kn : 0;   // last iter: harmless reload of kb=0
            // issue next-iteration loads before this iteration's WMMAs
            v16bf na0 = frag_row(arow, kc, half),  na1 = frag_row(arow, kc + 32, half);
            v16bf nb00 = frag_row(w0, kc, half),   nb01 = frag_row(w0, kc + 32, half);
            v16bf nb10 = frag_row(w1, kc, half),   nb11 = frag_row(w1, kc + 32, half);
            a00 = wmma_bf16(fa0, fb00, a00);
            a10 = wmma_bf16(fa0, fb10, a10);
            a01 = wmma_bf16(fa1, fb01, a01);
            a11 = wmma_bf16(fa1, fb11, a11);
            fa0 = na0; fa1 = na1;
            fb00 = nb00; fb01 = nb01; fb10 = nb10; fb11 = nb11;
        }
        v8f acc0 = a00 + a01;
        v8f acc1 = a10 + a11;
        float bb0 = bias[n0 + l];
        float bb1 = bias[n0 + 16 + l];
#pragma unroll
        for (int r = 0; r < 8; ++r) {
            int m = r + 8 * half; // batch index
            long base = ((long)m * TSTEPS + mt) * VOo;
            out[base + n0 + l] = acc0[r] + bb0;
            out[base + n0 + 16 + l] = acc1[r] + bb1;
        }
        __syncthreads();
    }
}

// ---------- workspace layout (bytes) ----------
#define O_WIH_E 0UL
#define O_WHH_E (O_WIH_E + 4096UL*512*2)
#define O_WIH_D (O_WHH_E + 4096UL*1024*2)
#define O_WHH_D (O_WIH_D + 4096UL*512*2)
#define O_WA    (O_WHH_D + 4096UL*1024*2)
#define O_UA    (O_WA    + 1024UL*1024*2)
#define O_FCW   (O_UA    + 1024UL*1024*2)
#define O_BIAS_E (O_FCW  + 32000UL*2048*2)
#define O_BIAS_D (O_BIAS_E + 4096UL*4)
#define O_XENC  (O_BIAS_D + 4096UL*4)
#define O_XDEC  (O_XENC + 64UL*16*512*2)
#define O_HBF0  (O_XDEC + 64UL*16*512*2)
#define O_HBF1  (O_HBF0 + 16UL*1024*2)
#define O_CF    (O_HBF1 + 16UL*1024*2)
#define O_ENCF  (O_CF   + 16UL*1024*4)
#define O_ENCB  (O_ENCF + 16UL*64*1024*4)
#define O_UE    (O_ENCB + 16UL*64*1024*2)
#define O_QF    (O_UE   + 16UL*64*1024*4)
#define O_CAT   (O_QF   + 16UL*1024*4)

extern "C" void kernel_launch(void* const* d_in, const int* in_sizes, int n_in,
                              void* d_out, int out_size, void* d_ws, size_t ws_size,
                              hipStream_t stream) {
    (void)in_sizes; (void)n_in; (void)out_size; (void)ws_size;
    const int*   src     = (const int*)d_in[0];
    const int*   tgt     = (const int*)d_in[1];
    const float* enc_emb = (const float*)d_in[2];
    const float* dec_emb = (const float*)d_in[3];
    const float* eWih    = (const float*)d_in[4];
    const float* eWhh    = (const float*)d_in[5];
    const float* eBih    = (const float*)d_in[6];
    const float* eBhh    = (const float*)d_in[7];
    const float* dWih    = (const float*)d_in[8];
    const float* dWhh    = (const float*)d_in[9];
    const float* dBih    = (const float*)d_in[10];
    const float* dBhh    = (const float*)d_in[11];
    const float* Wa      = (const float*)d_in[12];
    const float* ba      = (const float*)d_in[13];
    const float* Ua      = (const float*)d_in[14];
    const float* bu      = (const float*)d_in[15];
    const float* va      = (const float*)d_in[16];
    const float* bv      = (const float*)d_in[17];
    const float* fcW     = (const float*)d_in[18];
    const float* fcb     = (const float*)d_in[19];
    float* out = (float*)d_out;
    char* ws = (char*)d_ws;

    __bf16* wih_e = (__bf16*)(ws + O_WIH_E);
    __bf16* whh_e = (__bf16*)(ws + O_WHH_E);
    __bf16* wih_d = (__bf16*)(ws + O_WIH_D);
    __bf16* whh_d = (__bf16*)(ws + O_WHH_D);
    __bf16* wa_bf = (__bf16*)(ws + O_WA);
    __bf16* ua_bf = (__bf16*)(ws + O_UA);
    __bf16* fcw_bf = (__bf16*)(ws + O_FCW);
    float*  bias_e = (float*)(ws + O_BIAS_E);
    float*  bias_d = (float*)(ws + O_BIAS_D);
    __bf16* x_enc = (__bf16*)(ws + O_XENC);
    __bf16* x_dec = (__bf16*)(ws + O_XDEC);
    __bf16* hbf[2] = { (__bf16*)(ws + O_HBF0), (__bf16*)(ws + O_HBF1) };
    float*  c_f   = (float*)(ws + O_CF);
    float*  enc_f = (float*)(ws + O_ENCF);
    __bf16* enc_b = (__bf16*)(ws + O_ENCB);
    float*  ue_f  = (float*)(ws + O_UE);
    float*  q_f   = (float*)(ws + O_QF);
    __bf16* cat   = (__bf16*)(ws + O_CAT);

    // weight conversions (f32 -> bf16, L2-resident working set afterwards)
    cvt_kernel<<<1024, 256, 0, stream>>>(eWih, wih_e, 4096L*512/4);
    cvt_kernel<<<1024, 256, 0, stream>>>(eWhh, whh_e, 4096L*1024/4);
    cvt_kernel<<<1024, 256, 0, stream>>>(dWih, wih_d, 4096L*512/4);
    cvt_kernel<<<1024, 256, 0, stream>>>(dWhh, whh_d, 4096L*1024/4);
    cvt_kernel<<<1024, 256, 0, stream>>>(Wa, wa_bf, 1024L*1024/4);
    cvt_kernel<<<1024, 256, 0, stream>>>(Ua, ua_bf, 1024L*1024/4);
    cvt_kernel<<<4096, 256, 0, stream>>>(fcW, fcw_bf, 32000L*2048/4);
    bias_sum_kernel<<<16, 256, 0, stream>>>(eBih, eBhh, bias_e, 4096);
    bias_sum_kernel<<<16, 256, 0, stream>>>(dBih, dBhh, bias_d, 4096);
    hipMemsetAsync(hbf[0], 0, 16L*1024*2, stream);
    hipMemsetAsync(c_f,    0, 16L*1024*4, stream);

    // embeddings -> bf16, layout [step][batch][E]
    embed_kernel<<<(64*16*128 + 255)/256, 256, 0, stream>>>(src, 64, enc_emb, x_enc, 64);
    embed_kernel<<<(63*16*128 + 255)/256, 256, 0, stream>>>(tgt, 64, dec_emb, x_dec, 63);

    // encoder scan
    int cur = 0;
    for (int s = 0; s < 64; ++s) {
        lstm_step_kernel<<<64, 128, 0, stream>>>(
            x_enc + (long)s*16*Ee, hbf[cur], wih_e, whh_e, bias_e, c_f, hbf[cur^1],
            enc_f + (long)s*Hh, 64L*Hh, enc_b + (long)s*Hh, 64L*Hh);
        cur ^= 1;
    }

    // Ue = enc_out @ Ua^T + bu   (M=1024, N=1024, K=1024)
    gemm16_kernel<<<dim3(16, 64), 128, 0, stream>>>(enc_b, Hh, ua_bf, Hh, bu, ue_f, Hh);

    // decoder scan
    for (int t = 0; t < TSTEPS; ++t) {
        lstm_step_kernel<<<64, 128, 0, stream>>>(
            x_dec + (long)t*16*Ee, hbf[cur], wih_d, whh_d, bias_d, c_f, hbf[cur^1],
            nullptr, 0, cat + (long)t*16*2048, 2048);
        cur ^= 1;
        // q = h @ Wa^T + ba
        gemm16_kernel<<<dim3(16, 1), 128, 0, stream>>>(hbf[cur], Hh, wa_bf, Hh, ba, q_f, Hh);
        attention_kernel<<<16, 256, 0, stream>>>(q_f, ue_f, va, bv, enc_f,
                                                 cat + (long)t*16*2048);
    }

    // batched output projection, reads fc_W(bf16, 131MB, L2-resident) with M-reuse
    fc_gemm_kernel<<<dim3(125, 7), 256, 0, stream>>>(cat, fcw_bf, fcb, out);
}
